// L1OutUB_28114855919757
// MI455X (gfx1250) — matrix operand order repaired and checked by reading, not compiled, not run on hardware
//
#include <hip/hip_runtime.h>
#include <hip/hip_bf16.h>
#include <math.h>

typedef __attribute__((ext_vector_type(16))) _Float16 v16h;
typedef __attribute__((ext_vector_type(8)))  float    v8f;

#define DIMD 512
#define H2D  512
#define BD   384
#define RBLK 192

// ---------------------------------------------------------------------------
// Pack f32 row-major [M,K] -> f16 A-fragments (16x32 tiles, CDNA5 WMMA layout)
// Tile (tm,tk) occupies 512 f16 at ((tm*Kt+tk)*512); element = lane*16+e.
// lanes 0-15: M=lane, e 0..7 -> K 0..7, e 8..15 -> K 16..23
// lanes 16-31: M=lane-16, e 0..7 -> K 8..15, e 8..15 -> K 24..31
// ---------------------------------------------------------------------------
__global__ void pack_a_f16frag(const float* __restrict__ src,
                               _Float16* __restrict__ dst, int M, int K) {
  int t = blockIdx.x * blockDim.x + threadIdx.x;
  if (t >= M * K) return;
  int Kt = K >> 5;
  int tile = t >> 9, r = t & 511;
  int lane = r >> 4, e = r & 15;
  int tm = tile / Kt, tk = tile - tm * Kt;
  int m = lane & 15, half = lane >> 4;
  int k = (tk << 5) + ((e >> 3) << 4) + (half << 3) + (e & 7);
  dst[t] = (_Float16)src[(size_t)(tm * 16 + m) * K + k];
}

// ---------------------------------------------------------------------------
// Pack f32 row-major [K,N] -> f16 B-fragments (32x16 tiles).
// Tile (tk,tn) at ((tk*Nt+tn)*512); lane: n=lane&15, half=lane>>4; e: K=(e>>3)*16+half*8+(e&7).
// ---------------------------------------------------------------------------
__global__ void pack_b_f16frag(const float* __restrict__ src,
                               _Float16* __restrict__ dst, int K, int N) {
  int t = blockIdx.x * blockDim.x + threadIdx.x;
  if (t >= K * N) return;
  int Nt = N >> 4;
  int tile = t >> 9, r = t & 511;
  int lane = r >> 4, e = r & 15;
  int tk = tile / Nt, tn = tile - tk * Nt;
  int n = lane & 15, half = lane >> 4;
  int k = (tk << 5) + ((e >> 3) << 4) + (half << 3) + (e & 7);
  dst[t] = (_Float16)src[(size_t)k * N + tn * 16 + n];
}

// ---------------------------------------------------------------------------
// WMMA GEMM: C[M,N] = A[M,K] x B[K,N] + bias (optional ReLU), f32 out.
// One 16x16 tile per wave32; K consumed 32/step via v_wmma_f32_16x16x32_f16.
// ---------------------------------------------------------------------------
__global__ void gemm_wmma_f16(const _Float16* __restrict__ Apk,
                              const _Float16* __restrict__ Bpk,
                              const float* __restrict__ bias,
                              float* __restrict__ C,
                              int M, int N, int K, int do_relu) {
  int wid  = blockIdx.x * (blockDim.x >> 5) + (threadIdx.x >> 5);
  int lane = threadIdx.x & 31;
  int Mt = M >> 4, Nt = N >> 4, Kt = K >> 5;
  if (wid >= Mt * Nt) return;                 // wave-uniform
  int tm = wid / Nt, tn = wid - tm * Nt;

  const _Float16* ap = Apk + (size_t)tm * Kt * 512 + lane * 16;
  const _Float16* bp = Bpk + (size_t)tn * 512 + lane * 16;
  size_t bstride = (size_t)Nt * 512;

  v8f c = {};
  for (int tk = 0; tk < Kt; ++tk) {
    v16h a = *(const v16h*)ap;  ap += 512;
    v16h b = *(const v16h*)bp;  bp += bstride;
    c = __builtin_amdgcn_wmma_f32_16x16x32_f16(
        /*neg_a=*/false, a, /*neg_b=*/false, b,
        /*c_mod=*/(short)0, c, /*reuse_a=*/false, /*reuse_b=*/false);
  }

  int n = lane & 15, hi = lane >> 4;
  int col = tn * 16 + n;
  float bv = bias[col];
#pragma unroll
  for (int r2 = 0; r2 < 8; ++r2) {
    int row = tm * 16 + r2 + hi * 8;          // C/D layout: VGPR r -> M=r (+8 for hi lanes)
    float v = c[r2] + bv;
    if (do_relu) v = fmaxf(v, 0.0f);
    C[(size_t)row * N + col] = v;
  }
}

// Column moments of y: Sy[d] = sum_k y[k,d], Sy2[d] = sum_k y[k,d]^2.
__global__ void col_moments(const float* __restrict__ y,
                            float* __restrict__ Sy, float* __restrict__ Sy2) {
  int d = blockIdx.x * blockDim.x + threadIdx.x;
  if (d >= DIMD) return;
  float s = 0.f, s2 = 0.f;
  for (int k = 0; k < BD; ++k) {
    float v = y[(size_t)k * DIMD + d];
    s += v; s2 += v * v;
  }
  Sy[d] = s; Sy2[d] = s2;
}

// Fused O(B*D) reduction: positive-sum and all_probs-sum partials per block.
__global__ void reduce_main(const float* __restrict__ mu,
                            const float* __restrict__ lvraw,
                            const float* __restrict__ y,
                            const float* __restrict__ Sy,
                            const float* __restrict__ Sy2,
                            float* __restrict__ pos_part,
                            float* __restrict__ all_part) {
  __shared__ float sp[256];
  __shared__ float sa[256];
  int tid = threadIdx.x;
  const float Bf = (float)BD;
  float accp = 0.f, acca = 0.f;
  for (int idx = blockIdx.x * blockDim.x + tid; idx < BD * DIMD;
       idx += gridDim.x * blockDim.x) {
    int d = idx & (DIMD - 1);
    float lv = tanhf(lvraw[idx]);
    float iv = expf(-lv);
    float m  = mu[idx];
    float dy = m - y[idx];
    accp += -dy * dy * 0.5f * iv - 0.5f * lv;
    float q = Sy2[d] - 2.f * Sy[d] * m + Bf * m * m;   // sum_k (y[k,d]-m)^2
    acca += -0.5f * iv * q - 0.5f * Bf * lv;
  }
  sp[tid] = accp; sa[tid] = acca;
  __syncthreads();
  for (int s = 128; s > 0; s >>= 1) {
    if (tid < s) { sp[tid] += sp[tid + s]; sa[tid] += sa[tid + s]; }
    __syncthreads();
  }
  if (tid == 0) { pos_part[blockIdx.x] = sp[0]; all_part[blockIdx.x] = sa[0]; }
}

__global__ void finalize(const float* __restrict__ pos_part,
                         const float* __restrict__ all_part,
                         float* __restrict__ out) {
  __shared__ float sp[256];
  __shared__ float sa[256];
  int tid = threadIdx.x;
  sp[tid] = (tid < RBLK) ? pos_part[tid] : 0.f;
  sa[tid] = (tid < RBLK) ? all_part[tid] : 0.f;
  __syncthreads();
  for (int s = 128; s > 0; s >>= 1) {
    if (tid < s) { sp[tid] += sp[tid + s]; sa[tid] += sa[tid + s]; }
    __syncthreads();
  }
  if (tid == 0) {
    float meanpos = sp[0] / (float)BD;
    float meanall = sa[0] / ((float)BD * (float)BD);
    float delta = log1pf(expf(-20.f) / (float)(BD - 1));  // logsumexp mask residue
    out[0] = meanpos - meanall - delta;                   // == upper == forward value
  }
}

extern "C" void kernel_launch(void* const* d_in, const int* in_sizes, int n_in,
                              void* d_out, int out_size, void* d_ws, size_t ws_size,
                              hipStream_t stream) {
  const float* x    = (const float*)d_in[0];
  const float* y    = (const float*)d_in[1];
  const float* w1mu = (const float*)d_in[2];
  const float* b1mu = (const float*)d_in[3];
  const float* w2mu = (const float*)d_in[4];
  const float* b2mu = (const float*)d_in[5];
  const float* w1lv = (const float*)d_in[6];
  const float* b1lv = (const float*)d_in[7];
  const float* w2lv = (const float*)d_in[8];
  const float* b2lv = (const float*)d_in[9];
  float* out = (float*)d_out;

  char* ws = (char*)d_ws;
  size_t off = 0;
  auto alloc = [&](size_t bytes) { char* p = ws + off; off += (bytes + 255) & ~(size_t)255; return p; };

  _Float16* xA    = (_Float16*)alloc((size_t)BD * DIMD * 2);
  _Float16* w1mB  = (_Float16*)alloc((size_t)DIMD * H2D * 2);
  _Float16* w1lB  = (_Float16*)alloc((size_t)DIMD * H2D * 2);
  _Float16* w2mB  = (_Float16*)alloc((size_t)H2D * DIMD * 2);
  _Float16* w2lB  = (_Float16*)alloc((size_t)H2D * DIMD * 2);
  float*    hmu   = (float*)alloc((size_t)BD * H2D * 4);
  float*    hlv   = (float*)alloc((size_t)BD * H2D * 4);
  _Float16* hmuA  = (_Float16*)alloc((size_t)BD * H2D * 2);
  _Float16* hlvA  = (_Float16*)alloc((size_t)BD * H2D * 2);
  float*    muO   = (float*)alloc((size_t)BD * DIMD * 4);
  float*    lvO   = (float*)alloc((size_t)BD * DIMD * 4);
  float*    Sy    = (float*)alloc(DIMD * 4);
  float*    Sy2   = (float*)alloc(DIMD * 4);
  float*    posP  = (float*)alloc(RBLK * 4);
  float*    allP  = (float*)alloc(RBLK * 4);
  (void)ws_size; (void)in_sizes; (void)n_in; (void)out_size;

  const int PT = 256;
  // Pack inputs to WMMA fragment layouts (f32 -> f16)
  pack_a_f16frag<<<(BD * DIMD) / PT, PT, 0, stream>>>(x, xA, BD, DIMD);
  pack_b_f16frag<<<(DIMD * H2D) / PT, PT, 0, stream>>>(w1mu, w1mB, DIMD, H2D);
  pack_b_f16frag<<<(DIMD * H2D) / PT, PT, 0, stream>>>(w1lv, w1lB, DIMD, H2D);
  pack_b_f16frag<<<(H2D * DIMD) / PT, PT, 0, stream>>>(w2mu, w2mB, H2D, DIMD);
  pack_b_f16frag<<<(H2D * DIMD) / PT, PT, 0, stream>>>(w2lv, w2lB, H2D, DIMD);

  // Layer 1: relu(x @ w1 + b1)  — 24x32 = 768 tiles, 8 waves/block -> 96 blocks
  int gemm_blocks = ((BD / 16) * (H2D / 16)) / 8;
  gemm_wmma_f16<<<gemm_blocks, 256, 0, stream>>>(xA, w1mB, b1mu, hmu, BD, H2D, DIMD, 1);
  gemm_wmma_f16<<<gemm_blocks, 256, 0, stream>>>(xA, w1lB, b1lv, hlv, BD, H2D, DIMD, 1);

  // Repack hidden activations as A fragments
  pack_a_f16frag<<<(BD * H2D) / PT, PT, 0, stream>>>(hmu, hmuA, BD, H2D);
  pack_a_f16frag<<<(BD * H2D) / PT, PT, 0, stream>>>(hlv, hlvA, BD, H2D);

  // Layer 2: h @ w2 + b2 (mu raw, logvar raw — tanh applied in reduction)
  gemm_wmma_f16<<<gemm_blocks, 256, 0, stream>>>(hmuA, w2mB, b2mu, muO, BD, DIMD, H2D, 0);
  gemm_wmma_f16<<<gemm_blocks, 256, 0, stream>>>(hlvA, w2lB, b2lv, lvO, BD, DIMD, H2D, 0);

  // Moments of y, then fused O(B*D) reduction of positive & all_probs sums
  col_moments<<<DIMD / PT, PT, 0, stream>>>(y, Sy, Sy2);
  reduce_main<<<RBLK, 256, 0, stream>>>(muO, lvO, y, Sy, Sy2, posP, allP);
  finalize<<<1, 256, 0, stream>>>(posP, allP, out);
}
